// AutoPruneNet_53635551592612
// MI455X (gfx1250) — matrix operand order repaired and checked by reading, not compiled, not run on hardware
//
#include <hip/hip_runtime.h>

typedef _Float16 h16;
typedef __attribute__((ext_vector_type(16))) _Float16 v16h;
typedef __attribute__((ext_vector_type(8)))  _Float16 v8h;
typedef __attribute__((ext_vector_type(8)))  float    v8f;
typedef __attribute__((ext_vector_type(4)))  float    v4f;

#define TT      64
#define BBATCH  2048
#define TBROWS  (TT * BBATCH)       // 131072
#define OBSD    512
#define H1D     400
#define H2D     300
#define COREDIM 302

// Layer-1 tiling: N = 400 = 25 tiles of 16, K = 512 = 16 blocks of 32
#define N1T 25
#define K1B 16
// Layer-2 tiling: N padded to 304 = 19 tiles, K padded to 416 = 13 blocks
#define N2T 19
#define K2B 13
#define K2PAD 416
#define CORE_STRIDE 304

#define W1P_ELEMS (N1T * K1B * 512)   // 204800 f16
#define W2P_ELEMS (N2T * K2B * 512)   // 126464 f16

#define T1ELEMS (K1B * 512)           // 8192 halves / 16 KB per layer-1 tile
#define T2ELEMS (K2B * 512)           // 6656 halves / 13 KB per layer-2 tile

#define ACT1_HALVES (8 * 16 * K2PAD)  // 53248 halves
#define STAGE_HALVES 8192             // one buffer (max of T1/T2)

#define WMMA16(a, b, c) \
  __builtin_amdgcn_wmma_f32_16x16x32_f16(false, (a), false, (b), (short)0, (c), false, false)

// ---------------------------------------------------------------------------
// Repack W1 [400,512] f32 into f16 B-fragment-major layout.
// B fragment (32x16 f16, wave32): lane n holds column N = n%16;
//   half h (0..15) of the lane's v16h is K = kb*32 + (h&7) + 8*(lane>=16) + 16*(h>=8).
// Storage: idx = (nt*K1B + kb)*512 + lane*16 + h.
// ---------------------------------------------------------------------------
__global__ void pack_w1(const float* __restrict__ W1, h16* __restrict__ W1p) {
  int idx = blockIdx.x * 256 + threadIdx.x;
  if (idx >= W1P_ELEMS) return;
  int hh   = idx & 15;
  int lane = (idx >> 4) & 31;
  int frag = idx >> 9;              // nt*K1B + kb
  int kb   = frag & (K1B - 1);
  int nt   = frag >> 4;
  int n = nt * 16 + (lane & 15);
  int k = kb * 32 + (hh & 7) + ((lane >> 4) << 3) + ((hh >> 3) << 4);
  W1p[idx] = (h16)W1[n * OBSD + k];
}

// W2 [300,400] f32 -> padded (304x416) f16 fragment layout (zeros in pad).
__global__ void pack_w2(const float* __restrict__ W2, h16* __restrict__ W2p) {
  int idx = blockIdx.x * 256 + threadIdx.x;
  if (idx >= W2P_ELEMS) return;
  int hh   = idx & 15;
  int lane = (idx >> 4) & 31;
  int frag = idx >> 9;              // nt*K2B + kb
  int kb   = frag % K2B;
  int nt   = frag / K2B;
  int n = nt * 16 + (lane & 15);
  int k = kb * 32 + (hh & 7) + ((lane >> 4) << 3) + ((hh >> 3) << 4);
  float v = (n < H2D && k < H1D) ? W2[n * H1D + k] : 0.0f;
  W2p[idx] = (h16)v;
}

// ---------------------------------------------------------------------------
// Fused MLP: 8 waves per block, each wave owns 16 rows end-to-end.
// Weight tiles are staged into LDS ONCE PER WORKGROUP (8x traffic reduction)
// with a double-buffered pipeline: while tile nt is consumed from buf[cur],
// all 256 threads pull tile nt+1 global->regs, dump regs->buf[cur^1], barrier.
// corebuf aliases act1 (act1 dead once A2 fragments are in registers).
// LDS: 106.5 KB act1 + 32 KB staging = 139 KB -> 2 WGs per WGP.
// ---------------------------------------------------------------------------
__launch_bounds__(256)
__global__ void fused_net(const float* __restrict__ frame,
                          const float* __restrict__ reward,
                          const int*   __restrict__ last_action,
                          const float* __restrict__ eps,
                          const h16*   __restrict__ W1p,
                          const h16*   __restrict__ W2p,
                          const float* __restrict__ b1,
                          const float* __restrict__ b2,
                          const float* __restrict__ Wp,
                          const float* __restrict__ bp,
                          const float* __restrict__ Wb,
                          const float* __restrict__ bb,
                          float* __restrict__ out)
{
  extern __shared__ h16 smem[];
  const int tid   = threadIdx.x;
  const int wave  = tid >> 5;
  const int lane  = tid & 31;
  const int laneM = lane & 15;
  const int hi    = lane >> 4;      // 0 for lanes 0-15, 1 for lanes 16-31
  const int rowBase = blockIdx.x * 128 + wave * 16;

  h16* act1       = smem + wave * (16 * K2PAD);
  h16* corebuf    = act1;           // aliased: act1 dead after A2 load
  h16* stage_base = smem + ACT1_HALVES;  // two STAGE_HALVES buffers follow

  // ---- Load frame tile (16 rows x 512) once, straight into WMMA A-fragment
  //      register layout (f32 -> f16 convert in flight).
  v16h A[K1B];
  const float* fr = frame + (size_t)(rowBase + laneM) * OBSD;
  #pragma unroll
  for (int kb = 0; kb < K1B; ++kb) {
    const int base0 = kb * 32 + hi * 8;
    v4f x0 = *(const v4f*)(fr + base0);
    v4f x1 = *(const v4f*)(fr + base0 + 4);
    v4f x2 = *(const v4f*)(fr + base0 + 16);
    v4f x3 = *(const v4f*)(fr + base0 + 20);
    v16h a;
    #pragma unroll
    for (int j = 0; j < 4; ++j) {
      a[j]      = (h16)x0[j];
      a[4 + j]  = (h16)x1[j];
      a[8 + j]  = (h16)x2[j];
      a[12 + j] = (h16)x3[j];
    }
    A[kb] = a;
  }

  // ---- Stage layer-1 tile 0 into buf0 (each thread moves 32 halves) ----
  {
    const h16* src = W1p + tid * 32;
    v16h s0 = *(const v16h*)(src);
    v16h s1 = *(const v16h*)(src + 16);
    *(v16h*)(stage_base + tid * 32)      = s0;
    *(v16h*)(stage_base + tid * 32 + 16) = s1;
  }
  __syncthreads();
  int cur = 0;

  // ---- Layer 1: 25 N-tiles, K = 16 WMMA blocks, bias + ReLU -> LDS f16 ----
  for (int nt = 0; nt < N1T; ++nt) {
    // Prefetch next tile (layer-1 nt+1, or layer-2 tile 0 at the seam).
    const int nextElems = (nt + 1 < N1T) ? T1ELEMS : T2ELEMS;
    const h16* nsrc = (nt + 1 < N1T) ? (W1p + (size_t)(nt + 1) * T1ELEMS)
                                     : W2p;
    v16h s0 = {}, s1 = {};
    const bool doStage = (tid * 32) < nextElems;
    if (doStage) {
      s0 = *(const v16h*)(nsrc + tid * 32);
      s1 = *(const v16h*)(nsrc + tid * 32 + 16);
    }

    // Compute tile nt from LDS (two independent accumulator chains).
    const h16* wbuf = stage_base + cur * STAGE_HALVES + lane * 16;
    v8f acc0 = {}, acc1 = {};
    #pragma unroll
    for (int kb = 0; kb < K1B; kb += 2) {
      v16h bA = *(const v16h*)(wbuf + kb * 512);
      v16h bB = *(const v16h*)(wbuf + (kb + 1) * 512);
      acc0 = WMMA16(A[kb],     bA, acc0);
      acc1 = WMMA16(A[kb + 1], bB, acc1);
    }
    const float bias = b1[nt * 16 + laneM];
    #pragma unroll
    for (int r = 0; r < 8; ++r) {
      float v = acc0[r] + acc1[r] + bias;
      v = v > 0.0f ? v : 0.0f;
      act1[(r + 8 * hi) * K2PAD + nt * 16 + laneM] = (h16)v;
    }

    // Commit prefetched tile to the other buffer, then sync.
    if (doStage) {
      h16* dst = stage_base + (cur ^ 1) * STAGE_HALVES + tid * 32;
      *(v16h*)(dst)      = s0;
      *(v16h*)(dst + 16) = s1;
    }
    __syncthreads();
    cur ^= 1;
  }

  // Zero K padding (cols 400..415) for layer-2 accumulation.
  #pragma unroll
  for (int r = 0; r < 8; ++r)
    act1[(r + 8 * hi) * K2PAD + H1D + laneM] = (h16)0.0f;

  // ---- Load layer-2 A fragments from LDS (intra-wave, no barrier needed) --
  v16h A2[K2B];
  const h16* arow = act1 + laneM * K2PAD;
  #pragma unroll
  for (int kb = 0; kb < K2B; ++kb) {
    const int base0 = kb * 32 + hi * 8;
    v8h lo  = *(const v8h*)(arow + base0);
    v8h hi8 = *(const v8h*)(arow + base0 + 16);
    v16h a;
    #pragma unroll
    for (int j = 0; j < 8; ++j) { a[j] = lo[j]; a[8 + j] = hi8[j]; }
    A2[kb] = a;
  }

  // ---- Layer 2: 19 N-tiles, K = 13 WMMA blocks, bias + ReLU -> core f16 ---
  for (int nt = 0; nt < N2T; ++nt) {
    const int nextElems = (nt + 1 < N2T) ? T2ELEMS : 0;
    const h16* nsrc = W2p + (size_t)(nt + 1) * T2ELEMS;
    v16h s0 = {}, s1 = {};
    const bool doStage = (tid * 32) < nextElems;
    if (doStage) {
      s0 = *(const v16h*)(nsrc + tid * 32);
      s1 = *(const v16h*)(nsrc + tid * 32 + 16);
    }

    const h16* wbuf = stage_base + cur * STAGE_HALVES + lane * 16;
    v8f acc0 = {}, acc1 = {};
    #pragma unroll
    for (int kb = 0; kb + 1 < K2B; kb += 2) {
      v16h bA = *(const v16h*)(wbuf + kb * 512);
      v16h bB = *(const v16h*)(wbuf + (kb + 1) * 512);
      acc0 = WMMA16(A2[kb],     bA, acc0);
      acc1 = WMMA16(A2[kb + 1], bB, acc1);
    }
    {   // odd tail: kb = 12
      v16h bA = *(const v16h*)(wbuf + (K2B - 1) * 512);
      acc0 = WMMA16(A2[K2B - 1], bA, acc0);
    }
    const int n2 = nt * 16 + laneM;
    if (n2 < H2D) {
      const float bias = b2[n2];
      #pragma unroll
      for (int r = 0; r < 8; ++r) {
        float v = acc0[r] + acc1[r] + bias;
        v = v > 0.0f ? v : 0.0f;
        corebuf[(r + 8 * hi) * CORE_STRIDE + n2] = (h16)v;
      }
    }

    if (doStage) {
      h16* dst = stage_base + (cur ^ 1) * STAGE_HALVES + tid * 32;
      *(v16h*)(dst)      = s0;
      *(v16h*)(dst + 16) = s1;
    }
    __syncthreads();
    cur ^= 1;
  }

  // ---- Heads: all 32 lanes. Row = laneM; each 16-lane half covers 150 of
  //      the 300 hidden dims; cross-half combine via shuffle-xor(16). ----
  {
    const int grow  = rowBase + laneM;
    const int dBase = hi * 150;
    const h16* crow = corebuf + laneM * CORE_STRIDE;

    float p0 = 0.0f, p1 = 0.0f, pb = 0.0f;
    for (int d = 0; d < 150; ++d) {
      const int dd = dBase + d;
      float c = (float)crow[dd];
      p0 += c * Wp[dd];
      p1 += c * Wp[COREDIM + dd];
      pb += c * Wb[dd];
    }
    p0 += __shfl_xor(p0, 16, 32);
    p1 += __shfl_xor(p1, 16, 32);
    pb += __shfl_xor(pb, 16, 32);

    if (hi == 0) {
      const float cr = fminf(fmaxf(reward[grow], -1.0f), 1.0f);
      const float la = (float)last_action[grow];
      float a0 = p0 + bp[0] + cr * Wp[300] + la * Wp[301];
      float a1 = p1 + bp[1] + cr * Wp[COREDIM + 300] + la * Wp[COREDIM + 301];
      float ab = pb + bb[0] + cr * Wb[300] + la * Wb[301];

      const float mu = 1.0f / (1.0f + __expf(-a0));
      const float sg = 1.0f / (1.0f + __expf(-a1));
      const float action = mu + sg * eps[grow];

      float* o = out + (size_t)grow * 4;
      o[0] = mu;
      o[1] = sg;
      o[2] = ab;
      o[3] = action;
    }
  }
}

extern "C" void kernel_launch(void* const* d_in, const int* in_sizes, int n_in,
                              void* d_out, int out_size, void* d_ws, size_t ws_size,
                              hipStream_t stream) {
  const float* frame       = (const float*)d_in[0];
  const float* reward      = (const float*)d_in[1];
  const int*   last_action = (const int*)d_in[2];
  const float* eps         = (const float*)d_in[3];
  const float* W1          = (const float*)d_in[4];
  const float* b1          = (const float*)d_in[5];
  const float* W2          = (const float*)d_in[6];
  const float* b2          = (const float*)d_in[7];
  const float* Wp          = (const float*)d_in[8];
  const float* bp          = (const float*)d_in[9];
  const float* Wb          = (const float*)d_in[10];
  const float* bb          = (const float*)d_in[11];
  float* out = (float*)d_out;

  h16* W1p = (h16*)d_ws;
  h16* W2p = W1p + W1P_ELEMS;     // 409600 B offset, 32B aligned

  pack_w1<<<(W1P_ELEMS + 255) / 256, 256, 0, stream>>>(W1, W1p);
  pack_w2<<<(W2P_ELEMS + 255) / 256, 256, 0, stream>>>(W2, W2p);

  const int blocks = TBROWS / 128;  // 1024 workgroups, 8 waves each
  const size_t smem_bytes =
      (size_t)(ACT1_HALVES + 2 * STAGE_HALVES) * sizeof(h16);  // 139264 B
  fused_net<<<blocks, 256, smem_bytes, stream>>>(frame, reward, last_action,
                                                 eps, W1p, W2p, b1, b2, Wp, bp,
                                                 Wb, bb, out);
}
